// WindowMHSA1D_33758442946720
// MI455X (gfx1250) — matrix operand order repaired and checked by reading, not compiled, not run on hardware
//
#include <hip/hip_runtime.h>

// ---------------------------------------------------------------------------
// Swin-1D windowed MHSA for MI455X (gfx1250, wave32, WMMA bf16 16x16x32).
// Bandwidth-bound (~0.8GB min traffic @ 23.3TB/s); bf16 WMMA makes GEMM
// compute negligible. Layouts chosen so every WMMA fragment is contiguous
// per-lane 16B loads per the CDNA5 VGPR layout tables.
// ---------------------------------------------------------------------------

typedef __attribute__((ext_vector_type(16))) __bf16 v16bf;
typedef __attribute__((ext_vector_type(8)))  __bf16 v8bf;
typedef __attribute__((ext_vector_type(8)))  float  v8f;

#define WMMA_BF16(a, b, c) \
  __builtin_amdgcn_wmma_f32_16x16x32_bf16(false, (a), false, (b), (short)0, (c), false, false)

// A fragment (16x32 bf16): lane holds row (l&15); K chunks {sel*8..} and {16+sel*8..}
static __device__ __forceinline__ v16bf load_a_frag(const __bf16* p, int sel) {
  union { v16bf v; v8bf h[2]; } u;
  u.h[0] = *(const v8bf*)(p + sel * 8);
  u.h[1] = *(const v8bf*)(p + 16 + sel * 8);
  return u.v;
}
// B fragment (32x16 bf16): lane holds col (l&15); K chunks {sel*8..} and {16+sel*8..}
static __device__ __forceinline__ v16bf load_b_frag(const __bf16* p, int sel) {
  union { v16bf v; v8bf h[2]; } u;
  u.h[0] = *(const v8bf*)(p + sel * 8);
  u.h[1] = *(const v8bf*)(p + 16 + sel * 8);
  return u.v;
}

// ---------------- kernel 0: roll(+32) + fp32->bf16 convert of x -------------
__global__ void k_roll_cvt(const float* __restrict__ x, __bf16* __restrict__ xb) {
  long i = (long)blockIdx.x * blockDim.x + threadIdx.x;   // 16*4096*768 elems
  int  d = (int)(i % 768);
  long t = i / 768;
  int  m = (int)(t & 4095);
  long b = t >> 12;
  long src = ((b << 12) + ((m - 32) & 4095)) * 768 + d;   // rolled token
  xb[i] = (__bf16)x[src];
}

// ---------------- kernel 0b: plain fp32->bf16 convert (weights) -------------
__global__ void k_cvt(const float* __restrict__ in, __bf16* __restrict__ out, int n) {
  int i = blockIdx.x * blockDim.x + threadIdx.x;
  if (i < n) out[i] = (__bf16)in[i];
}

// ---------------- kernel 1: QKV GEMM  C[65536,2304] = Xb @ Wqkv^T + b -------
// block = 256 thr (8 waves, 4Mx2N), block tile 128x128, wave tile 32x64.
// Epilogue scatters into Q[k] row-major [win][h][tok][64] and V transposed
// [win][h][d][tok] so attention's B fragments are contiguous.
__global__ void k_qkv_gemm(const __bf16* __restrict__ xb,
                           const __bf16* __restrict__ wb,
                           const float*  __restrict__ qkvb,
                           __bf16* __restrict__ qb,
                           __bf16* __restrict__ kb,
                           __bf16* __restrict__ vtb) {
  const int lane = threadIdx.x & 31;
  const int wv   = threadIdx.x >> 5;
  const int wm = wv & 3, wn = wv >> 2;
  const int m0 = blockIdx.y * 128 + wm * 32;
  const int n0 = blockIdx.x * 128 + wn * 64;
  const int lr = lane & 15, sel = lane >> 4;

  v8f acc[2][4] = {};
  const __bf16* arow0 = xb + (long)(m0 + lr) * 768;
  const __bf16* arow1 = arow0 + (long)16 * 768;
  const __bf16* bcol0 = wb + (long)(n0 + lr) * 768;

#pragma unroll 4
  for (int k0 = 0; k0 < 768; k0 += 32) {
    v16bf a0 = load_a_frag(arow0 + k0, sel);
    v16bf a1 = load_a_frag(arow1 + k0, sel);
#pragma unroll
    for (int j = 0; j < 4; ++j) {
      v16bf bf = load_b_frag(bcol0 + (long)j * 16 * 768 + k0, sel);
      acc[0][j] = WMMA_BF16(a0, bf, acc[0][j]);
      acc[1][j] = WMMA_BF16(a1, bf, acc[1][j]);
    }
  }

#pragma unroll
  for (int j = 0; j < 4; ++j) {
    int n = n0 + j * 16 + lr;
    float bias = qkvb[n];
    int region = n / 768;                // 0=Q 1=K 2=V
    int nn = n - region * 768;
    int h = nn >> 6, d = nn & 63;
#pragma unroll
    for (int i = 0; i < 2; ++i) {
#pragma unroll
      for (int r = 0; r < 8; ++r) {
        int row = m0 + i * 16 + r + sel * 8;     // global token (rolled coords)
        int win = row >> 6, tok = row & 63;
        __bf16 o = (__bf16)(acc[i][j][r] + bias);
        long hb = (long)(win * 12 + h) * 64;
        if (region == 0)      qb [(hb + tok) * 64 + d]   = o;
        else if (region == 1) kb [(hb + tok) * 64 + d]   = o;
        else                  vtb[(hb + d)   * 64 + tok] = o;
      }
    }
  }
}

// ---------------- kernel 2: per (window,head) attention ---------------------
// 128 threads = 4 waves; wave wv owns S/Y rows [wv*16, wv*16+16).
__global__ void k_attn(const __bf16* __restrict__ qb,
                       const __bf16* __restrict__ kb,
                       const __bf16* __restrict__ vtb,
                       const float*  __restrict__ relb,
                       __bf16* __restrict__ yb) {
  __shared__ alignas(32) float  Sf[64][68];
  __shared__ alignas(32) __bf16 P [64][80];
  __shared__ float rinv[64];

  const int wh  = blockIdx.x;
  const int win = wh / 12, h = wh - win * 12;
  const __bf16* q  = qb  + (long)wh * 64 * 64;
  const __bf16* k  = kb  + (long)wh * 64 * 64;
  const __bf16* vt = vtb + (long)wh * 64 * 64;
  const int lane = threadIdx.x & 31, wv = threadIdx.x >> 5;
  const int lr = lane & 15, sel = lane >> 4;

  // ---- S = (q @ k^T) * scale ----
  {
    const __bf16* arow = q + (wv * 16 + lr) * 64;
    v8f acc[4] = {};
#pragma unroll
    for (int ks = 0; ks < 64; ks += 32) {
      v16bf a = load_a_frag(arow + ks, sel);
#pragma unroll
      for (int nt = 0; nt < 4; ++nt) {
        v16bf bf = load_b_frag(k + (nt * 16 + lr) * 64 + ks, sel);
        acc[nt] = WMMA_BF16(a, bf, acc[nt]);
      }
    }
#pragma unroll
    for (int nt = 0; nt < 4; ++nt)
#pragma unroll
      for (int r = 0; r < 8; ++r)
        Sf[wv * 16 + r + sel * 8][nt * 16 + lr] = acc[nt][r] * 0.125f;
  }
  __syncthreads();

  // ---- softmax rows (unnormalized exp; fold 1/sum into PV epilogue) ----
  if (threadIdx.x < 64) {
    int i = threadIdx.x;
    const float* rb = relb + h * 127 + 63 - i;   // rel = j - i + 63
    float mx = -1e30f;
    for (int j = 0; j < 64; ++j) mx = fmaxf(mx, Sf[i][j] + rb[j]);
    float sum = 0.f;
    for (int j = 0; j < 64; ++j) {
      float e = __expf(Sf[i][j] + rb[j] - mx);
      sum += e;
      P[i][j] = (__bf16)e;
    }
    rinv[i] = 1.0f / sum;
  }
  __syncthreads();

  // ---- Y = P @ V (V pre-transposed -> contiguous B fragments) ----
  {
    const __bf16* arow = &P[wv * 16 + lr][0];
    v8f acc[4] = {};
#pragma unroll
    for (int ks = 0; ks < 64; ks += 32) {
      v16bf a = load_a_frag(arow + ks, sel);
#pragma unroll
      for (int nt = 0; nt < 4; ++nt) {
        v16bf bf = load_b_frag(vt + (nt * 16 + lr) * 64 + ks, sel);
        acc[nt] = WMMA_BF16(a, bf, acc[nt]);
      }
    }
#pragma unroll
    for (int nt = 0; nt < 4; ++nt)
#pragma unroll
      for (int r = 0; r < 8; ++r) {
        int row = wv * 16 + r + sel * 8;
        float v = acc[nt][r] * rinv[row];
        yb[((long)win * 64 + row) * 768 + h * 64 + nt * 16 + lr] = (__bf16)v;
      }
  }
}

// ---------------- kernel 3: Proj GEMM + inverse roll on store ---------------
__global__ void k_proj_gemm(const __bf16* __restrict__ yb,
                            const __bf16* __restrict__ wb,
                            const float*  __restrict__ pb,
                            float* __restrict__ out) {
  const int lane = threadIdx.x & 31;
  const int wv   = threadIdx.x >> 5;
  const int wm = wv & 3, wn = wv >> 2;
  const int m0 = blockIdx.y * 128 + wm * 32;
  const int n0 = blockIdx.x * 128 + wn * 64;
  const int lr = lane & 15, sel = lane >> 4;

  v8f acc[2][4] = {};
  const __bf16* arow0 = yb + (long)(m0 + lr) * 768;
  const __bf16* arow1 = arow0 + (long)16 * 768;
  const __bf16* bcol0 = wb + (long)(n0 + lr) * 768;

#pragma unroll 4
  for (int k0 = 0; k0 < 768; k0 += 32) {
    v16bf a0 = load_a_frag(arow0 + k0, sel);
    v16bf a1 = load_a_frag(arow1 + k0, sel);
#pragma unroll
    for (int j = 0; j < 4; ++j) {
      v16bf bf = load_b_frag(bcol0 + (long)j * 16 * 768 + k0, sel);
      acc[0][j] = WMMA_BF16(a0, bf, acc[0][j]);
      acc[1][j] = WMMA_BF16(a1, bf, acc[1][j]);
    }
  }

#pragma unroll
  for (int j = 0; j < 4; ++j) {
    int n = n0 + j * 16 + lr;
    float bias = pb[n];
#pragma unroll
    for (int i = 0; i < 2; ++i) {
#pragma unroll
      for (int r = 0; r < 8; ++r) {
        int row = m0 + i * 16 + r + sel * 8;     // rolled token index
        long b  = row >> 12;
        int  mr = row & 4095;
        int  orow = (mr - 32) & 4095;            // inverse roll
        out[((b << 12) + orow) * (long)768 + n] = acc[i][j][r] + bias;
      }
    }
  }
}

// ---------------------------------------------------------------------------
extern "C" void kernel_launch(void* const* d_in, const int* in_sizes, int n_in,
                              void* d_out, int out_size, void* d_ws, size_t ws_size,
                              hipStream_t stream) {
  const float* x      = (const float*)d_in[0];
  const float* qkv_w  = (const float*)d_in[1];
  const float* qkv_b  = (const float*)d_in[2];
  const float* proj_w = (const float*)d_in[3];
  const float* proj_b = (const float*)d_in[4];
  const float* rel_b  = (const float*)d_in[5];
  float* out = (float*)d_out;

  char* ws = (char*)d_ws;
  const size_t SZ_ACT = (size_t)65536 * 768 * 2;          // 100,663,296 B
  __bf16* xb    = (__bf16*)(ws + 0 * SZ_ACT);             // rolled x, bf16
  __bf16* qb    = (__bf16*)(ws + 1 * SZ_ACT);             // [win][h][tok][64]
  __bf16* kb    = (__bf16*)(ws + 2 * SZ_ACT);             // [win][h][tok][64]
  __bf16* vtb   = (__bf16*)(ws + 3 * SZ_ACT);             // [win][h][d][tok]
  __bf16* wqkv  = (__bf16*)(ws + 4 * SZ_ACT);             // 3,538,944 B
  __bf16* wproj = (__bf16*)(ws + 4 * SZ_ACT + (size_t)2304 * 768 * 2);
  __bf16* ybuf  = xb;   // xb dead after QKV GEMM -> alias as attention output

  k_roll_cvt<<<196608, 256, 0, stream>>>(x, xb);
  k_cvt<<<6912, 256, 0, stream>>>(qkv_w, wqkv, 2304 * 768);
  k_cvt<<<2304, 256, 0, stream>>>(proj_w, wproj, 768 * 768);

  k_qkv_gemm<<<dim3(18, 512), 256, 0, stream>>>(xb, wqkv, qkv_b, qb, kb, vtb);
  k_attn<<<12288, 128, 0, stream>>>(qb, kb, vtb, rel_b, ybuf);
  k_proj_gemm<<<dim3(6, 512), 256, 0, stream>>>(ybuf, wproj, proj_b, out);
}
